// TritonRaggedDeiT_78898549227595
// MI455X (gfx1250) — matrix operand order repaired and checked by reading, not compiled, not run on hardware
//
#include <hip/hip_runtime.h>
#include <math.h>

// ---------------------------------------------------------------------------
// Ragged DeiT forward for MI455X (gfx1250, wave32, WMMA bf16 16x16x32 + TDM)
// Double-buffered Tensor-Data-Mover pipeline feeding the WMMA GEMMs.
// ---------------------------------------------------------------------------

#define D      768
#define H      12
#define HD     64
#define L      6
#define FF     3072
#define NC     1000
#define QKVN   (3 * D)     // 2304
#define BATCH  128
#define SPAD   224         // 197 padded to multiple of 32
#define TOT    (BATCH * SPAD)

#define USE_TDM 1          // Tensor Data Mover staging of GEMM tiles

typedef __attribute__((ext_vector_type(16))) __bf16        v16bf;
typedef __attribute__((ext_vector_type(8)))  float         v8f;
typedef __attribute__((ext_vector_type(4)))  unsigned int  u32x4;
typedef __attribute__((ext_vector_type(8)))  unsigned int  u32x8;

union ABFrag {
    v16bf        v;
    unsigned int u[8];
};

__device__ __forceinline__ unsigned short f32_to_bf16(float f) {
    unsigned u = __float_as_uint(f);
    unsigned r = (u + 0x7FFFu + ((u >> 16) & 1u)) >> 16;
    return (unsigned short)r;
}

__device__ __forceinline__ v8f wmma_bf16(const v16bf& a, const v16bf& b, v8f c) {
    // D = A(16x32 bf16) * B(32x16 bf16) + C(16x16 f32)
    return __builtin_amdgcn_wmma_f32_16x16x32_bf16(
        false, a, false, b, (short)0, c, false, false);
}

#if USE_TDM
// ---------------------------------------------------------------------------
// Tensor Data Mover: DMA a 2D bf16 tile (tile_h x tile_w) from a row-major
// matrix in global memory into LDS (contiguous [tile_h][tile_w]).
// D# layout per CDNA5 ISA. Tracked by TENSORcnt.
// ---------------------------------------------------------------------------
__device__ __forceinline__ void tdm_load_tile_2d(
    const void* gaddr, unsigned lds_off_bytes,
    unsigned tensor_w, unsigned tensor_h,          // extents for OOB zero-fill
    unsigned tile_w, unsigned tile_h,
    unsigned long long row_stride_elems) {
    unsigned long long ga = (unsigned long long)(size_t)gaddr;
    u32x4 g0;
    g0[0] = 1u;                                    // count=1, is_restore=0
    g0[1] = lds_off_bytes;                         // lds_addr
    g0[2] = (unsigned)ga;                          // global_addr[31:0]
    g0[3] = (unsigned)(ga >> 32) | (2u << 30);     // global_addr[56:32] | type=2
    u32x8 g1;
    g1[0] = 1u << 16;                              // data_size=1 (2 bytes)
    g1[1] = (tensor_w & 0xFFFFu) << 16;            // tensor_dim0[15:0]
    g1[2] = (tensor_w >> 16) | ((tensor_h & 0xFFFFu) << 16); // dim0 hi | dim1 lo
    g1[3] = (tensor_h >> 16) | (tile_w << 16);     // dim1 hi | tile_dim0
    g1[4] = tile_h;                                // tile_dim1 (tile_dim2=0)
    g1[5] = (unsigned)row_stride_elems;            // tensor_dim0_stride[31:0]
    g1[6] = (unsigned)(row_stride_elems >> 32) & 0xFFFFu; // stride0 hi (stride1=0)
    g1[7] = 0u;
    u32x4 z = {0u, 0u, 0u, 0u};
    asm volatile("tensor_load_to_lds %0, %1, %2, %3"
                 :: "s"(g0), "s"(g1), "s"(z), "s"(z)
                 : "memory");
}

__device__ __forceinline__ void tdm_wait0() {
#if __has_builtin(__builtin_amdgcn_s_wait_tensorcnt)
    __builtin_amdgcn_s_wait_tensorcnt(0);
#else
    asm volatile("s_wait_tensorcnt 0x0" ::: "memory");
#endif
}
#endif  // USE_TDM

// ---------------------------------------------------------------------------
// fp32 -> bf16 weight conversion
// ---------------------------------------------------------------------------
__global__ void cvt_bf16_kernel(const float* __restrict__ src,
                                unsigned short* __restrict__ dst, size_t n) {
    size_t i = (size_t)blockIdx.x * blockDim.x + threadIdx.x;
    if (i < n) dst[i] = f32_to_bf16(src[i]);
}

// ---------------------------------------------------------------------------
// Gather packed ragged tokens into dense padded [TOT, D] fp32
// ---------------------------------------------------------------------------
__global__ void gather_pad_kernel(const float* __restrict__ packed,
                                  const int* __restrict__ cu,
                                  float* __restrict__ xf) {
    int row = blockIdx.x;                // 0..TOT-1
    int b = row / SPAD, p = row % SPAD;
    int start = cu[b];
    int len   = cu[b + 1] - start;
    float* dst = xf + (size_t)row * D;
    if (p < len) {
        const float* src = packed + (size_t)(start + p) * D;
        for (int i = threadIdx.x; i < D; i += blockDim.x) dst[i] = src[i];
    } else {
        for (int i = threadIdx.x; i < D; i += blockDim.x) dst[i] = 0.0f;
    }
}

// ---------------------------------------------------------------------------
// LayerNorm over D=768, one wave32 per row, bf16 output.
// ---------------------------------------------------------------------------
__global__ __launch_bounds__(256)
void ln_kernel(const float* __restrict__ x, const float* __restrict__ w,
               const float* __restrict__ bias, unsigned short* __restrict__ out,
               int nrows, int rowStride) {
    const int lane = threadIdx.x & 31;
    const int wave = threadIdx.x >> 5;
    const int row  = blockIdx.x * 8 + wave;
    if (row >= nrows) return;
    const float* xr = x + (size_t)row * rowStride * D;

    float vals[D / 32];
    float s = 0.0f;
#pragma unroll
    for (int i = 0; i < D / 32; ++i) { vals[i] = xr[lane + 32 * i]; s += vals[i]; }
#pragma unroll
    for (int off = 16; off > 0; off >>= 1) s += __shfl_xor(s, off, 32);
    const float mu = s * (1.0f / D);

    float vsum = 0.0f;
#pragma unroll
    for (int i = 0; i < D / 32; ++i) { float d = vals[i] - mu; vsum += d * d; }
#pragma unroll
    for (int off = 16; off > 0; off >>= 1) vsum += __shfl_xor(vsum, off, 32);
    const float rstd = rsqrtf(vsum * (1.0f / D) + 1e-6f);

    unsigned short* orow = out + (size_t)row * D;
#pragma unroll
    for (int i = 0; i < D / 32; ++i) {
        int c = lane + 32 * i;
        orow[c] = f32_to_bf16((vals[i] - mu) * rstd * w[c] + bias[c]);
    }
}

// ---------------------------------------------------------------------------
// bf16 WMMA GEMM: C[M,N] = A[M,K] @ B[K,N] (+ bias), epilogues:
// EPI 0: bf16 out (bias)        EPI 1: bf16 out (bias + exact GELU)
// EPI 2: fp32 residual add      EPI 3: fp32 out (bias)
// Block tile 128x128x32; 8 waves = 4(M) x 2(N); wave tile 32x64
// => 8 WMMAs / K-step per wave, B frags reused across both A frags.
// Double-buffered TDM pipeline: tile i+1 DMA'd (TENSORcnt) into the alternate
// LDS buffer while tile i is consumed by the WMMA pipe; one barrier per step.
// A as [128][32] row-major, B as [32][128] row-major.
// ---------------------------------------------------------------------------
#define BLK_M 128
#define BLK_N 128
#define BK    32

template <int EPI>
__global__ __launch_bounds__(256)
void gemm_bf16_wmma(const unsigned short* __restrict__ A,
                    const unsigned short* __restrict__ B,
                    const float* __restrict__ bias,
                    unsigned short* __restrict__ Cb,
                    float* __restrict__ Cf,
                    int M, int N, int K) {
    __shared__ alignas(16) unsigned short As[2][BLK_M * BK];   // [m][k]
    __shared__ alignas(16) unsigned short Bs[2][BK * BLK_N];   // [k][n]

    const int tid  = threadIdx.x;
    const int lane = tid & 31;
    const int wave = tid >> 5;
    const int wr   = wave >> 1;        // 0..3  (M direction)
    const int wc   = wave & 1;         // 0..1  (N direction)
    const int m0   = blockIdx.y * BLK_M;
    const int n0   = blockIdx.x * BLK_N;
    const int nk   = K / BK;

    v8f zero = {0.f, 0.f, 0.f, 0.f, 0.f, 0.f, 0.f, 0.f};
    v8f acc[2][4] = {{zero, zero, zero, zero}, {zero, zero, zero, zero}};

#if USE_TDM
    const unsigned ldsA0 = (unsigned)(size_t)(const void*)&As[0][0];
    const unsigned ldsA1 = (unsigned)(size_t)(const void*)&As[1][0];
    const unsigned ldsB0 = (unsigned)(size_t)(const void*)&Bs[0][0];
    const unsigned ldsB1 = (unsigned)(size_t)(const void*)&Bs[1][0];

    // issue the pair of tile DMAs for K-step i into buffer (i & 1)
    auto issue = [&](int i) {
        int k0 = i * BK;
        unsigned la = (i & 1) ? ldsA1 : ldsA0;
        unsigned lb = (i & 1) ? ldsB1 : ldsB0;
        // A tile: rows m0..m0+127, cols k0..k0+31 of [M,K]
        tdm_load_tile_2d(A + (size_t)m0 * K + k0, la,
                         (unsigned)(K - k0), (unsigned)(M - m0),
                         BK, BLK_M, (unsigned long long)K);
        // B tile: rows k0..k0+31, cols n0..n0+127 of [K,N]
        tdm_load_tile_2d(B + (size_t)k0 * N + n0, lb,
                         (unsigned)(N - n0), (unsigned)(K - k0),
                         BLK_N, BK, (unsigned long long)N);
    };

    if (wave == 0) {                   // prologue: fill buffer 0
        issue(0);
        tdm_wait0();
    }
    __syncthreads();
#endif

    for (int i = 0; i < nk; ++i) {
        const int stage = i & 1;
#if USE_TDM
        if (wave == 0 && i + 1 < nk)
            issue(i + 1);              // overlap next DMA with this compute
        const unsigned short* Asb = &As[stage][0];
        const unsigned short* Bsb = &Bs[stage][0];
#else
        const int k0 = i * BK;
        // manual staging fallback (same LDS layout, buffer 0 only)
#pragma unroll
        for (int j = 0; j < 8; ++j) {           // A: 2048 dword pairs
            int pp  = tid + 256 * j;
            int row = pp >> 4;
            int kk  = (pp & 15) << 1;
            int gm  = m0 + row;
            unsigned int val = 0u;
            if (gm < M) val = *(const unsigned int*)(A + (size_t)gm * K + k0 + kk);
            *(unsigned int*)(&As[0][row * BK + kk]) = val;
        }
#pragma unroll
        for (int j = 0; j < 16; ++j) {          // B: 4096 elements
            int e  = tid + 256 * j;
            int kk = e >> 7;
            int nn = e & 127;
            int gn = n0 + nn;
            Bs[0][kk * BLK_N + nn] = (gn < N) ? B[(size_t)(k0 + kk) * N + gn]
                                              : (unsigned short)0;
        }
        __syncthreads();
        const unsigned short* Asb = &As[0][0];
        const unsigned short* Bsb = &Bs[0][0];
        (void)stage;
#endif

        // --- A fragments (contiguous k = 16h + s layout, same as B), wave rows wr*32..+31 ---
        ABFrag af[2];
        {
            const int m  = lane & 15;
            const int kb = (lane < 16) ? 0 : 16;
#pragma unroll
            for (int s = 0; s < 2; ++s) {
                const unsigned short* ap = Asb + (wr * 32 + s * 16 + m) * BK;
#pragma unroll
                for (int v = 0; v < 8; ++v) {
                    int koff = kb + 2 * v;
                    af[s].u[v] = *(const unsigned int*)(ap + koff);
                }
            }
        }
        // --- B fragments + 8 WMMAs ---
        const int n   = lane & 15;
        const int kb2 = (lane < 16) ? 0 : 16;
#pragma unroll
        for (int t = 0; t < 4; ++t) {
            ABFrag bf;
            const unsigned short* bp = Bsb + wc * 64 + t * 16 + n;
#pragma unroll
            for (int v = 0; v < 8; ++v) {
                unsigned lo = bp[(kb2 + 2 * v) * BLK_N];
                unsigned hi = bp[(kb2 + 2 * v + 1) * BLK_N];
                bf.u[v] = lo | (hi << 16);
            }
            acc[0][t] = wmma_bf16(af[0].v, bf.v, acc[0][t]);
            acc[1][t] = wmma_bf16(af[1].v, bf.v, acc[1][t]);
        }

#if USE_TDM
        if (wave == 0 && i + 1 < nk)
            tdm_wait0();               // next tile has landed in alt buffer
#endif
        __syncthreads();               // fence before reusing/advancing buffers
    }

    // --- epilogue (C layout: lanes 0-15 => M=r, lanes 16-31 => M=8+r) ---
    const int ncol = lane & 15;
    const int mofs = (lane < 16) ? 0 : 8;
#pragma unroll
    for (int s = 0; s < 2; ++s) {
#pragma unroll
        for (int t = 0; t < 4; ++t) {
            int gn = n0 + wc * 64 + t * 16 + ncol;
            if (gn >= N) continue;
            float bv = bias ? bias[gn] : 0.0f;
#pragma unroll
            for (int r = 0; r < 8; ++r) {
                int gm = m0 + wr * 32 + s * 16 + mofs + r;
                if (gm >= M) continue;
                float val = acc[s][t][r] + bv;
                size_t o = (size_t)gm * N + gn;
                if (EPI == 0) {
                    Cb[o] = f32_to_bf16(val);
                } else if (EPI == 1) {
                    float g = 0.5f * val * (1.0f + erff(val * 0.70710678118654752f));
                    Cb[o] = f32_to_bf16(g);
                } else if (EPI == 2) {
                    Cf[o] += val;
                } else {
                    Cf[o] = val;
                }
            }
        }
    }
}

// ---------------------------------------------------------------------------
// Attention: one wave per (16-query tile, head, batch).
// S = Q K^T * 0.125 (masked), softmax, O = P V. All matmuls via WMMA.
// qkv layout per token row: [3 * H * HD], s-major (q:0, k:768, v:1536).
// ---------------------------------------------------------------------------
__global__ __launch_bounds__(32)
void attn_kernel(const unsigned short* __restrict__ qkv,
                 const int* __restrict__ cu,
                 unsigned short* __restrict__ obuf) {
    __shared__ alignas(16) float          S[16][SPAD];
    __shared__ alignas(16) unsigned short P[16][SPAD];

    const int lane = threadIdx.x;
    const int qt   = blockIdx.x;   // 0..13 query tiles
    const int h    = blockIdx.y;   // head
    const int b    = blockIdx.z;   // batch
    const int len  = cu[b + 1] - cu[b];
    const size_t t0 = (size_t)b * SPAD;
    const float scale = 0.125f;    // 1/sqrt(64)

    v8f zero = {0.f, 0.f, 0.f, 0.f, 0.f, 0.f, 0.f, 0.f};

    // ---- Q fragments (2 K-steps over HD=64) straight from global ----
    ABFrag aq[2];
    {
        const int m  = lane & 15;
        const int kb = (lane < 16) ? 0 : 16;
        const unsigned short* qp = qkv + (t0 + qt * 16 + m) * QKVN + h * HD;
#pragma unroll
        for (int ds = 0; ds < 2; ++ds) {
#pragma unroll
            for (int v = 0; v < 8; ++v) {
                int koff = kb + 2 * v;
                aq[ds].u[v] = *(const unsigned int*)(qp + ds * 32 + koff);
            }
        }
    }

    // ---- scores S = Q K^T (K row-major == transposed-B layout) ----
    {
        const int n    = lane & 15;
        const int kb2  = (lane < 16) ? 0 : 16;
        const int mofs = (lane < 16) ? 0 : 8;
        for (int kt = 0; kt < SPAD / 16; ++kt) {
            v8f acc = zero;
            const unsigned short* kp =
                qkv + (t0 + kt * 16 + n) * QKVN + D + h * HD;
#pragma unroll
            for (int ds = 0; ds < 2; ++ds) {
                ABFrag bk;
#pragma unroll
                for (int v = 0; v < 8; ++v)
                    bk.u[v] = *(const unsigned int*)(kp + ds * 32 + kb2 + 2 * v);
                acc = wmma_bf16(aq[ds].v, bk.v, acc);
            }
            int col = kt * 16 + n;
            bool kvalid = col < len;
#pragma unroll
            for (int r = 0; r < 8; ++r)
                S[mofs + r][col] = kvalid ? acc[r] * scale : -1e30f;
        }
    }
    __syncthreads();

    // ---- softmax per query row (lanes 0..15 own one row each) ----
    if (lane < 16) {
        float mx = -3.0e38f;
        for (int c = 0; c < SPAD; ++c) mx = fmaxf(mx, S[lane][c]);
        float sum = 0.f;
        for (int c = 0; c < SPAD; ++c) {
            float e = __expf(S[lane][c] - mx);
            S[lane][c] = e;
            sum += e;
        }
        float inv = 1.0f / sum;
        for (int c = 0; c < SPAD; ++c)
            P[lane][c] = f32_to_bf16(S[lane][c] * inv);
    }
    __syncthreads();

    // ---- O = P @ V  (K-dim = SPAD keys, 7 WMMA K-steps) ----
    {
        const int m    = lane & 15;
        const int kbA  = (lane < 16) ? 0 : 16;
        const int n    = lane & 15;
        const int kb2  = (lane < 16) ? 0 : 16;
        const int mofs = (lane < 16) ? 0 : 8;
#pragma unroll
        for (int dt = 0; dt < 4; ++dt) {
            v8f acc = zero;
            for (int ks = 0; ks < SPAD / 32; ++ks) {
                ABFrag ap;
#pragma unroll
                for (int v = 0; v < 8; ++v) {
                    int koff = kbA + 2 * v;
                    ap.u[v] = *(const unsigned int*)(&P[m][ks * 32 + koff]);
                }
                ABFrag bv;
                const unsigned short* vp = qkv + 2 * D + h * HD + dt * 16 + n;
#pragma unroll
                for (int v = 0; v < 8; ++v) {
                    size_t key = t0 + ks * 32 + kb2 + 2 * v;
                    unsigned lo = vp[key * QKVN];
                    unsigned hi = vp[(key + 1) * QKVN];
                    bv.u[v] = lo | (hi << 16);
                }
                acc = wmma_bf16(ap.v, bv.v, acc);
            }
#pragma unroll
            for (int r = 0; r < 8; ++r) {
                size_t row = t0 + qt * 16 + mofs + r;
                obuf[row * D + h * HD + dt * 16 + n] = f32_to_bf16(acc[r]);
            }
        }
    }
}

// ---------------------------------------------------------------------------
// Host orchestration
// ---------------------------------------------------------------------------
extern "C" void kernel_launch(void* const* d_in, const int* in_sizes, int n_in,
                              void* d_out, int out_size, void* d_ws, size_t ws_size,
                              hipStream_t stream) {
    const float* packed  = (const float*)d_in[0];
    const int*   cu      = (const int*)d_in[1];
    const float* norm1_w = (const float*)d_in[2];
    const float* norm1_b = (const float*)d_in[3];
    const float* qkv_w   = (const float*)d_in[4];
    const float* qkv_b   = (const float*)d_in[5];
    const float* proj_w  = (const float*)d_in[6];
    const float* proj_b  = (const float*)d_in[7];
    const float* norm2_w = (const float*)d_in[8];
    const float* norm2_b = (const float*)d_in[9];
    const float* fc1_w   = (const float*)d_in[10];
    const float* fc1_b   = (const float*)d_in[11];
    const float* fc2_w   = (const float*)d_in[12];
    const float* fc2_b   = (const float*)d_in[13];
    const float* bnorm_w = (const float*)d_in[14];
    const float* bnorm_b = (const float*)d_in[15];
    const float* head_w  = (const float*)d_in[16];
    const float* head_b  = (const float*)d_in[17];
    float* out = (float*)d_out;

    // ---- workspace carve ----
    char*  ws  = (char*)d_ws;
    size_t off = 0;
    auto carve = [&](size_t bytes) -> void* {
        void* p = ws + off;
        off = (off + bytes + 255) & ~(size_t)255;
        return p;
    };
    float*          xf   = (float*)carve((size_t)TOT * D * 4);
    unsigned short* hb   = (unsigned short*)carve((size_t)TOT * D * 2);
    unsigned short* qkvb = (unsigned short*)carve((size_t)TOT * QKVN * 2);
    unsigned short* ob   = (unsigned short*)carve((size_t)TOT * D * 2);
    unsigned short* ffb  = (unsigned short*)carve((size_t)TOT * FF * 2);
    unsigned short* wq   = (unsigned short*)carve((size_t)L * D * QKVN * 2);
    unsigned short* wp   = (unsigned short*)carve((size_t)L * D * D * 2);
    unsigned short* w1   = (unsigned short*)carve((size_t)L * D * FF * 2);
    unsigned short* w2   = (unsigned short*)carve((size_t)L * FF * D * 2);
    unsigned short* wh   = (unsigned short*)carve((size_t)D * NC * 2);
    unsigned short* clsb = (unsigned short*)carve((size_t)BATCH * D * 2);

    auto cvt = [&](const float* s, unsigned short* d, size_t n) {
        cvt_bf16_kernel<<<(unsigned)((n + 255) / 256), 256, 0, stream>>>(s, d, n);
    };
    cvt(qkv_w,  wq, (size_t)L * D * QKVN);
    cvt(proj_w, wp, (size_t)L * D * D);
    cvt(fc1_w,  w1, (size_t)L * D * FF);
    cvt(fc2_w,  w2, (size_t)L * FF * D);
    cvt(head_w, wh, (size_t)D * NC);

    gather_pad_kernel<<<TOT, 256, 0, stream>>>(packed, cu, xf);

    auto gemm_grid = [](int M, int N) {
        return dim3((N + BLK_N - 1) / BLK_N, (M + BLK_M - 1) / BLK_M);
    };

    for (int l = 0; l < L; ++l) {
        ln_kernel<<<(TOT + 7) / 8, 256, 0, stream>>>(
            xf, norm1_w + l * D, norm1_b + l * D, hb, TOT, 1);
        gemm_bf16_wmma<0><<<gemm_grid(TOT, QKVN), 256, 0, stream>>>(
            hb, wq + (size_t)l * D * QKVN, qkv_b + (size_t)l * QKVN,
            qkvb, nullptr, TOT, QKVN, D);
        attn_kernel<<<dim3(SPAD / 16, H, BATCH), 32, 0, stream>>>(qkvb, cu, ob);
        gemm_bf16_wmma<2><<<gemm_grid(TOT, D), 256, 0, stream>>>(
            ob, wp + (size_t)l * D * D, proj_b + (size_t)l * D,
            nullptr, xf, TOT, D, D);
        ln_kernel<<<(TOT + 7) / 8, 256, 0, stream>>>(
            xf, norm2_w + l * D, norm2_b + l * D, hb, TOT, 1);
        gemm_bf16_wmma<1><<<gemm_grid(TOT, FF), 256, 0, stream>>>(
            hb, w1 + (size_t)l * D * FF, fc1_b + (size_t)l * FF,
            ffb, nullptr, TOT, FF, D);
        gemm_bf16_wmma<2><<<gemm_grid(TOT, D), 256, 0, stream>>>(
            ffb, w2 + (size_t)l * FF * D, fc2_b + (size_t)l * D,
            nullptr, xf, TOT, D, FF);
    }

    ln_kernel<<<(BATCH + 7) / 8, 256, 0, stream>>>(
        xf, bnorm_w, bnorm_b, clsb, BATCH, SPAD);
    gemm_bf16_wmma<3><<<gemm_grid(BATCH, NC), 256, 0, stream>>>(
        clsb, wh, head_b, nullptr, out, BATCH, NC, D);
}
